// AttentionDecoder_23467701305571
// MI455X (gfx1250) — matrix operand order repaired and checked by reading, not compiled, not run on hardware
//
#include <hip/hip_runtime.h>
#include <hip/hip_bf16.h>
#include <math.h>

// ---------------------------------------------------------------------------
// AttentionDecoder fused kernels for MI455X (gfx1250, wave32, WMMA 16x16x32 f16)
//
//   L=512, N=64, S=512, E=256, H=8, head_dim=32  (head_dim == WMMA K !)
//
//   1) proj_f16: Y = (X @ W^T + b) * scale  -> f16 via v_wmma_f32_16x16x32_f16
//   2) attn: block = (n, 16 l-rows); wave w owns s-cols [w*64, w*64+64) and
//      loops all 8 heads (1 WMMA per head per 16-col s-tile). Head-sums stay
//      in registers -> plain ds_store (no LDS atomics). Softmax uses a FIXED
//      reference m=10 (tanh clip bounds scores) -> no max tracking, no
//      per-tile shuffles; per-lane z partials reduced once at the end.
// ---------------------------------------------------------------------------

typedef __attribute__((ext_vector_type(16))) _Float16 v16h;
typedef __attribute__((ext_vector_type(8)))  _Float16 v8h;
typedef __attribute__((ext_vector_type(8)))  float    v8f;

#define LQ  512
#define SK  512
#define NB  64
#define EM  256
#define NH  8
#define SP  (SK + 4)   // padded LDS row: 516 floats -> halves land on disjoint banks

// ---------------------------------------------------------------------------
// Projection: Y[row][o] = (sum_e X[row][e] * W[o][e] + b[o]) * scale, f16 out.
// ---------------------------------------------------------------------------
__global__ __launch_bounds__(256) void proj_f16_kernel(
    const float* __restrict__ X, const float* __restrict__ W,
    const float* __restrict__ bias, _Float16* __restrict__ Y, float scale) {
  const int wave = threadIdx.x >> 5;
  const int lane = threadIdx.x & 31;
  const int hf   = lane >> 4;
  const int ln16 = lane & 15;

  const int row0 = blockIdx.x * 16;
  const int c0   = blockIdx.y * 128 + wave * 16;

  const float* xrow = X + (size_t)(row0 + ln16) * EM;   // A row (M = ln16)
  const int    ocol = c0 + ln16;                        // B/C col (N = ln16)
  const float* wrow = W + (size_t)ocol * EM;

  v8f c = {};
#pragma unroll
  for (int kk = 0; kk < EM; kk += 32) {
    // A 16x32 f16: lanes<16: h0-7 K=0..7, h8-15 K=16..23 | lanes>=16: +8
    v16h a;
    const int e0 = kk + (hf ? 8 : 0);
#pragma unroll
    for (int j = 0; j < 8; ++j) a[j]     = (_Float16)xrow[e0 + j];
#pragma unroll
    for (int j = 0; j < 8; ++j) a[8 + j] = (_Float16)xrow[e0 + 16 + j];
    // B 32x16 f16: col = lane%16; lanes<16: h0-7 K=0..7, h8-15 K=16..23 | lanes>=16: +8
    v16h b;
    const int eb = kk + (hf ? 8 : 0);
#pragma unroll
    for (int j = 0; j < 8; ++j) b[j]     = (_Float16)wrow[eb + j];
#pragma unroll
    for (int j = 0; j < 8; ++j) b[8 + j] = (_Float16)wrow[eb + 16 + j];
    c = __builtin_amdgcn_wmma_f32_16x16x32_f16(false, a, false, b,
                                               (short)0, c, false, false);
  }

  const float bv = bias[ocol];
#pragma unroll
  for (int r = 0; r < 8; ++r) {
    const int row = row0 + r + (hf ? 8 : 0);
    Y[(size_t)row * EM + ocol] = (_Float16)((c[r] + bv) * scale);
  }
}

// ---------------------------------------------------------------------------
// Fused attention. Grid: (L/16, N). Block: 256 = 8 waves; wave w owns
// s-columns [w*64, w*64+64) and computes all 8 heads for them.
// ---------------------------------------------------------------------------
__global__ __launch_bounds__(256) void attn_kernel(
    const _Float16* __restrict__ qh, const _Float16* __restrict__ kh,
    const unsigned char* __restrict__ mask, float* __restrict__ out) {
  __shared__ float         sums[16][SP];      // ~33 KB head-summed tanh scores
  __shared__ unsigned char mk[16][SK];        //  8 KB mask tile
  __shared__ float         zpart[NH][NH][16]; //  4 KB [wave][head][l-row]
  __shared__ float         lseS[NH][16];
  __shared__ float         mlse[16];

  const int n    = blockIdx.y;
  const int l0   = blockIdx.x * 16;
  const int tid  = threadIdx.x;
  const int wave = tid >> 5;
  const int lane = tid & 31;
  const int hf   = lane >> 4;
  const int ln16 = lane & 15;

  // ---- stage mask tile (16 contiguous rows of 512 B) ----
  {
    const uint4* msrc = reinterpret_cast<const uint4*>(mask + ((size_t)n * LQ + l0) * SK);
    uint4* mdst = reinterpret_cast<uint4*>(&mk[0][0]);
    for (int i = tid; i < 16 * SK / 16; i += 256) mdst[i] = msrc[i];
  }

  // ---- A fragments: full 16x256 q tile, one 16x32 frag per head ----
  const _Float16* qrow = qh + ((size_t)(l0 + ln16) * NB + n) * EM;
  const int d0 = hf ? 16 : 0;
  v16h a[NH];
#pragma unroll
  for (int h = 0; h < NH; ++h) {
    v8h lo = *reinterpret_cast<const v8h*>(qrow + h * 32 + d0);
    v8h hi = *reinterpret_cast<const v8h*>(qrow + h * 32 + d0 + 8);
#pragma unroll
    for (int j = 0; j < 8; ++j) { a[h][j] = lo[j]; a[h][8 + j] = hi[j]; }
  }

  float z[NH][8];
#pragma unroll
  for (int h = 0; h < NH; ++h)
#pragma unroll
    for (int r = 0; r < 8; ++r) z[h][r] = 0.f;

  __syncthreads();  // mask tile visible

  const size_t krowstride = (size_t)NB * EM;
  const size_t kcoloff    = (size_t)n * EM + (hf ? 16 : 0);

  for (int t = 0; t < 4; ++t) {
    const int st = wave * 64 + t * 16;
    const int s  = st + ln16;
    const _Float16* krow = kh + (size_t)s * krowstride + kcoloff;  // 512B/lane stream
    if (t < 3)
      __builtin_prefetch(kh + (size_t)(s + 16) * krowstride + kcoloff, 0, 3);

    // per-tile mask factors for this lane's column (1 = keep, 0 = masked)
    float mfac[8];
#pragma unroll
    for (int r = 0; r < 8; ++r) mfac[r] = mk[r + (hf ? 8 : 0)][s] ? 0.f : 1.f;

    float sum_r[8];
#pragma unroll
    for (int r = 0; r < 8; ++r) sum_r[r] = 0.f;

#pragma unroll
    for (int h = 0; h < NH; ++h) {
      // B 32x16 frag: k[s, h*32 + ...], 32B contiguous per lane
      v16h b = *reinterpret_cast<const v16h*>(krow + h * 32);
      v8f c = {};
      c = __builtin_amdgcn_wmma_f32_16x16x32_f16(false, a[h], false, b,
                                                 (short)0, c, false, false);
#pragma unroll
      for (int r = 0; r < 8; ++r) {
        // th = 10*tanh(sc) = 10 - 20/(exp(2sc)+1); clamp dodges exp overflow
        const float sc = fminf(fmaxf(c[r], -15.f), 15.f);
        const float e2 = __expf(2.f * sc);
        const float th = 10.f - 20.f * __builtin_amdgcn_rcpf(e2 + 1.f);
        sum_r[r] += th;                               // masked slots unused later
        z[h][r]  += mfac[r] * __expf(th - 10.f);      // fixed ref m=10, exact
      }
    }

    // conflict-free stores: 16 consecutive cols per half; halves bank-disjoint
#pragma unroll
    for (int r = 0; r < 8; ++r) sums[r + (hf ? 8 : 0)][s] = sum_r[r];
  }

  // ---- one-time lane reduction of z partials (within each 16-lane half) ----
#pragma unroll
  for (int h = 0; h < NH; ++h)
#pragma unroll
    for (int r = 0; r < 8; ++r) {
      float v = z[h][r];
      v += __shfl_xor(v, 1);
      v += __shfl_xor(v, 2);
      v += __shfl_xor(v, 4);
      v += __shfl_xor(v, 8);
      z[h][r] = v;
    }
  if (ln16 == 0) {
#pragma unroll
    for (int h = 0; h < NH; ++h)
#pragma unroll
      for (int r = 0; r < 8; ++r) zpart[wave][h][r + (hf ? 8 : 0)] = z[h][r];
  }
  __syncthreads();

  // ---- lse per (head, row), then per-row mean over heads ----
  if (tid < NH * 16) {
    const int h = tid >> 4, row = tid & 15;
    float zt = 0.f;
#pragma unroll
    for (int w = 0; w < NH; ++w) zt += zpart[w][h][row];
    lseS[h][row] = __logf(zt);
  }
  __syncthreads();
  if (tid < 16) {
    float acc = 0.f;
#pragma unroll
    for (int h = 0; h < NH; ++h) acc += lseS[h][tid];
    mlse[tid] = acc * 0.125f + 10.f;   // mean_h(10 + log z_h)
  }
  __syncthreads();

  // ---- out = sums/8 - mlse ; masked -> -inf (float4 coalesced writes) ----
  float* orow = out + ((size_t)n * LQ + l0) * SK;
  for (int i = tid; i < 16 * SK / 4; i += 256) {
    const int lr = i >> 7;               // 128 float4 per row
    const int s4 = (i & 127) << 2;
    const float4 sv = *reinterpret_cast<const float4*>(&sums[lr][s4]);
    const unsigned mb = *reinterpret_cast<const unsigned*>(&mk[lr][s4]);
    const float ml = mlse[lr];
    float4 o;
    o.x = (mb & 0x000000ffu) ? -INFINITY : sv.x * 0.125f - ml;
    o.y = (mb & 0x0000ff00u) ? -INFINITY : sv.y * 0.125f - ml;
    o.z = (mb & 0x00ff0000u) ? -INFINITY : sv.z * 0.125f - ml;
    o.w = (mb & 0xff000000u) ? -INFINITY : sv.w * 0.125f - ml;
    *reinterpret_cast<float4*>(&orow[(size_t)lr * SK + s4]) = o;
  }
}

// ---------------------------------------------------------------------------
extern "C" void kernel_launch(void* const* d_in, const int* in_sizes, int n_in,
                              void* d_out, int out_size, void* d_ws, size_t ws_size,
                              hipStream_t stream) {
  const float*         query = (const float*)d_in[0];          // (L,N,256)
  const float*         key   = (const float*)d_in[1];          // (S,N,256)
  const unsigned char* amask = (const unsigned char*)d_in[2];  // (N,L,S) bool
  const float*         Wq    = (const float*)d_in[3];
  const float*         bq    = (const float*)d_in[4];
  const float*         Wk    = (const float*)d_in[5];
  const float*         bk    = (const float*)d_in[6];
  float*               out   = (float*)d_out;

  _Float16* qh = (_Float16*)d_ws;                       // 16.78 MB
  _Float16* kh = qh + (size_t)LQ * NB * EM;             // +16.78 MB (ws >= 33.6MB)

  const float scaling = 0.17677669529663687f;           // 1/sqrt(32)

  dim3 pb(256);
  dim3 pg(LQ * NB / 16, 2);
  proj_f16_kernel<<<pg, pb, 0, stream>>>(query, Wq, bq, qh, scaling);
  proj_f16_kernel<<<pg, pb, 0, stream>>>(key,   Wk, bk, kh, 1.0f);

  dim3 ag(LQ / 16, NB);
  attn_kernel<<<ag, 256, 0, stream>>>(qh, kh, amask, out);
}
